// FR_12343736008794
// MI455X (gfx1250) — matrix operand order repaired and checked by reading, not compiled, not run on hardware
//
#include <hip/hip_runtime.h>

// ---------------------------------------------------------------------------
// MI455X (gfx1250) implementation.
// Heavy math: 4 GEMMs of 512x32768x1024 (~137 GFLOP) -> v_wmma_f32_16x16x32_bf16
// Double-buffered LDS staging (global_load_b128 of tile k+1 overlaps the WMMA
// block of tile k), native v_cvt f32->bf16, global_prefetch two tiles ahead.
// Everything softmax/gating-related stays fp32 scalar VALU.
// ws layout (floats):
//   [0          , 16777216)   x1x  fp32
//   [16777216   , 33554432)   x2x  fp32
//   [33554432   , +16384)     pooled1
//   [33570816   , +16384)     pooled2
//   [33587200   , +16384)     gate1 (sigmoid already applied)
//   [33603584   , +16384)     gate2
//   [33619968 .. )            RE bf16 [32][1024][1024]  (64 MiB)
// total ws required ~= 201.6 MB
// ---------------------------------------------------------------------------

typedef __attribute__((ext_vector_type(16))) __bf16       v16bf;
typedef __attribute__((ext_vector_type(8)))  float        v8f;
typedef __attribute__((ext_vector_type(4)))  float        v4f;
typedef __attribute__((ext_vector_type(4)))  unsigned int v4u;
typedef __attribute__((ext_vector_type(2)))  unsigned int v2u;

#define HWSZ 1024   // H*W
#define KDIM 1024   // concat channels
#define OC   512    // output channels per conv
#define NB   32     // batch
#define LDSTRIDE 40 // bf16 row stride: 80B rows -> 16B-aligned b128 frag reads

__device__ __forceinline__ unsigned short f2bf(float f) {
  return __builtin_bit_cast(unsigned short, (__bf16)f);   // native cvt (RNE)
}
__device__ __forceinline__ unsigned pk2bf(float a, float b) {
  return (unsigned)f2bf(a) | ((unsigned)f2bf(b) << 16);   // -> v_cvt_pk_bf16_f32
}

union Frag { v16bf v; v4u q[2]; };

// ---------------------------------------------------------------------------
// Dual-weight GEMM: out{A,B}[n][o][hw] = sum_c W{a,b}[o][c] * B[n][c][hw] + b[o]
// B source is either fp32 split (x1 | x2 channel concat) or a bf16 buffer.
// Grid: x = hw tile (8), y = m tile (8: 0-3 -> Wa, 4-7 -> Wb), z = n (32).
// Block: 256 threads = 8 waves (2 M x 4 N). Block tile 128x128, K step 32.
// ---------------------------------------------------------------------------
template <int BF16IN>
__global__ __launch_bounds__(256) void gemm_dual_kernel(
    const float* __restrict__ Wa, const float* __restrict__ ba, float* __restrict__ outA,
    const float* __restrict__ Wb, const float* __restrict__ bb, float* __restrict__ outB,
    const float* __restrict__ Bs1, const float* __restrict__ Bs2,
    const unsigned short* __restrict__ Bbf) {
  __shared__ unsigned short sA[2][128 * LDSTRIDE];   // [m][k]
  __shared__ unsigned short sB[2][128 * LDSTRIDE];   // transposed: [j(hw)][k]

  const int tid   = threadIdx.x;
  const int lane  = tid & 31;
  const int wave  = tid >> 5;
  const int waveM = wave >> 2;              // 0..1
  const int waveN = wave & 3;               // 0..3
  const int lrow  = lane & 15;
  const int lhi   = lane >> 4;

  const int n  = blockIdx.z;
  const int mt = blockIdx.y;
  const bool useB = (mt >= 4);
  const float* W    = useB ? Wb : Wa;
  const float* bias = useB ? bb : ba;
  float*       out  = useB ? outB : outA;
  const int m0 = (mt & 3) * 128;
  const int n0 = blockIdx.x * 128;

  // per-thread staging coordinates (fixed across K tiles)
  const int ar  = tid >> 3;                 // A row 0..31 (per pass: +32*p)
  const int ac4 = (tid & 7) * 4;            // A col group
  const int bk  = tid >> 5;                 // B k-row 0..7 (per pass: +8*p)
  const int bj4 = (tid & 31) * 4;           // B hw group

  v8f acc[4][2];
#pragma unroll
  for (int i = 0; i < 4; ++i)
#pragma unroll
    for (int j = 0; j < 2; ++j) acc[i][j] = (v8f)0.0f;

  v4f ra[4];          // staged A (fp32)
  v4f rbf[4];         // staged B (fp32 path)
  v2u rbu[4];         // staged B (bf16 path)

  // ---- tile loaders / LDS writers ------------------------------------------
  auto load_tile = [&](int kt) {
#pragma unroll
    for (int p = 0; p < 4; ++p) {
      ra[p] = *(const v4f*)(W + (size_t)(m0 + p * 32 + ar) * KDIM + kt * 32 + ac4);
      const int c = kt * 32 + p * 8 + bk;
      if (BF16IN) {
        rbu[p] = *(const v2u*)(Bbf + ((size_t)n * KDIM + c) * HWSZ + n0 + bj4);
      } else {
        const float* src = (c < 512)
            ? (Bs1 + ((size_t)n * 512 + c)         * HWSZ + n0 + bj4)
            : (Bs2 + ((size_t)n * 512 + (c - 512)) * HWSZ + n0 + bj4);
        rbf[p] = *(const v4f*)src;
      }
    }
    if (kt + 1 < KDIM / 32) {               // prefetch one more tile ahead
      __builtin_prefetch(W + (size_t)(m0 + ar) * KDIM + (kt + 1) * 32 + ac4, 0, 1);
      if (BF16IN)
        __builtin_prefetch(Bbf + ((size_t)n * KDIM + (kt + 1) * 32 + bk) * HWSZ + n0 + bj4, 0, 1);
    }
  };
  auto store_tile = [&](int buf) {
#pragma unroll
    for (int p = 0; p < 4; ++p) {
      v2u st;
      st.x = pk2bf(ra[p].x, ra[p].y);
      st.y = pk2bf(ra[p].z, ra[p].w);
      *(v2u*)(&sA[buf][(p * 32 + ar) * LDSTRIDE + ac4]) = st;
      const int kk = p * 8 + bk;
      unsigned short h0, h1, h2, h3;
      if (BF16IN) {
        h0 = (unsigned short)(rbu[p].x & 0xffff); h1 = (unsigned short)(rbu[p].x >> 16);
        h2 = (unsigned short)(rbu[p].y & 0xffff); h3 = (unsigned short)(rbu[p].y >> 16);
      } else {
        h0 = f2bf(rbf[p].x); h1 = f2bf(rbf[p].y);
        h2 = f2bf(rbf[p].z); h3 = f2bf(rbf[p].w);
      }
      sB[buf][(bj4 + 0) * LDSTRIDE + kk] = h0;
      sB[buf][(bj4 + 1) * LDSTRIDE + kk] = h1;
      sB[buf][(bj4 + 2) * LDSTRIDE + kk] = h2;
      sB[buf][(bj4 + 3) * LDSTRIDE + kk] = h3;
    }
  };

  // ---- prologue: stage tile 0 ----------------------------------------------
  load_tile(0);
  store_tile(0);
  __syncthreads();

  // ---- main K loop, double buffered ----------------------------------------
  for (int kt = 0; kt < KDIM / 32; ++kt) {
    const int buf = kt & 1;
    const bool more = (kt + 1 < KDIM / 32);
    if (more) load_tile(kt + 1);            // overlaps WMMA below

    Frag a[4], b[2];
#pragma unroll
    for (int wm = 0; wm < 4; ++wm) {
      const int m = waveM * 64 + wm * 16 + lrow;
      // lane<16: K = {0..7,16..23}; lane>=16: K = {8..15,24..31}
      a[wm].q[0] = *(const v4u*)(&sA[buf][m * LDSTRIDE + 8 * lhi]);
      a[wm].q[1] = *(const v4u*)(&sA[buf][m * LDSTRIDE + 16 + 8 * lhi]);
    }
#pragma unroll
    for (int wn = 0; wn < 2; ++wn) {
      const int j = waveN * 32 + wn * 16 + lrow;
      // lane<16: K = {0..7,16..23}; lane>=16: K = {8..15,24..31}
      b[wn].q[0] = *(const v4u*)(&sB[buf][j * LDSTRIDE + 8 * lhi]);
      b[wn].q[1] = *(const v4u*)(&sB[buf][j * LDSTRIDE + 16 + 8 * lhi]);
    }
#pragma unroll
    for (int wm = 0; wm < 4; ++wm)
#pragma unroll
      for (int wn = 0; wn < 2; ++wn)
        acc[wm][wn] = __builtin_amdgcn_wmma_f32_16x16x32_bf16(
            false, a[wm].v, false, b[wn].v, (short)0, acc[wm][wn], false, false);

    if (more) store_tile(buf ^ 1);
    __syncthreads();
  }

  // ---- epilogue: bias add, fp32 store --------------------------------------
#pragma unroll
  for (int wm = 0; wm < 4; ++wm)
#pragma unroll
    for (int wn = 0; wn < 2; ++wn) {
      const int obase = m0 + waveM * 64 + wm * 16 + 8 * lhi;
      const int hw    = n0 + waveN * 32 + wn * 16 + lrow;
#pragma unroll
      for (int r = 0; r < 8; ++r) {
        const int o = obase + r;
        out[((size_t)n * OC + o) * HWSZ + hw] = acc[wm][wn][r] + bias[o];
      }
    }
}

// ---------------------------------------------------------------------------
// pooled[n][c] = sum_hw softmax_hw(exp(x))*x  — one wave per (n,c) row.
// ---------------------------------------------------------------------------
__global__ __launch_bounds__(256) void pool_kernel(
    const float* __restrict__ X1X, const float* __restrict__ X2X,
    float* __restrict__ P1, float* __restrict__ P2) {
  const int lane = threadIdx.x & 31;
  const int wave = threadIdx.x >> 5;
  const float* X = blockIdx.y ? X2X : X1X;
  float*       P = blockIdx.y ? P2 : P1;
  const int row  = blockIdx.x * 8 + wave;          // 0 .. 16383
  const float* src = X + (size_t)row * HWSZ;

  float m = -1e30f, z = 0.f, w = 0.f;
  for (int i = lane; i < HWSZ; i += 32) {
    const float x = src[i];
    const float t = expf(x);                       // key of softmax(exp(x))
    if (t > m) { const float s = expf(m - t); z = z * s + 1.f; w = w * s + x; m = t; }
    else       { const float e = expf(t - m); z += e; w += e * x; }
  }
#pragma unroll
  for (int d = 16; d > 0; d >>= 1) {
    const float m2 = __shfl_xor(m, d, 32);
    const float z2 = __shfl_xor(z, d, 32);
    const float w2 = __shfl_xor(w, d, 32);
    const float nm = fmaxf(m, m2);
    const float sa = expf(m - nm), sb = expf(m2 - nm);
    z = z * sa + z2 * sb;
    w = w * sa + w2 * sb;
    m = nm;
  }
  if (lane == 0) P[row] = w / z;
}

// ---------------------------------------------------------------------------
// gate[n][o] = sigmoid(W2 (W1 pooled[n] + b1) + b2) — one block per (n,branch)
// ---------------------------------------------------------------------------
__global__ __launch_bounds__(256) void mlp_gate_kernel(
    const float* __restrict__ P1, const float* __restrict__ P2,
    const float* __restrict__ w1a, const float* __restrict__ b1a,
    const float* __restrict__ w2a, const float* __restrict__ b2a,
    const float* __restrict__ w1b, const float* __restrict__ b1b,
    const float* __restrict__ w2b, const float* __restrict__ b2b,
    float* __restrict__ G1, float* __restrict__ G2) {
  __shared__ float sp[512];
  __shared__ float sh[512];
  const int n = blockIdx.x;
  const int br = blockIdx.y;
  const float* P  = br ? P2 : P1;
  const float* w1 = br ? w1b : w1a;  const float* b1 = br ? b1b : b1a;
  const float* w2 = br ? w2b : w2a;  const float* b2 = br ? b2b : b2a;
  float* G = br ? G2 : G1;
  const int tid = threadIdx.x;

  for (int i = tid; i < 512; i += 256) sp[i] = P[(size_t)n * 512 + i];
  __syncthreads();
  for (int o = tid; o < 512; o += 256) {
    const float* wr = w1 + (size_t)o * 512;
    float s = b1[o];
    for (int c = 0; c < 512; ++c) s = fmaf(wr[c], sp[c], s);
    sh[o] = s;
  }
  __syncthreads();
  for (int o = tid; o < 512; o += 256) {
    const float* wr = w2 + (size_t)o * 512;
    float s = b2[o];
    for (int c = 0; c < 512; ++c) s = fmaf(wr[c], sh[c], s);
    G[(size_t)n * 512 + o] = 1.f / (1.f + expf(-s));
  }
}

// ---------------------------------------------------------------------------
// Per-pixel: channel-softmax spatial gates + residuals; emits RE (bf16 concat).
// Grid: x = hw chunk (4), y = n (32). One thread per pixel.
// ---------------------------------------------------------------------------
__global__ __launch_bounds__(256) void fuse_spatial_kernel(
    const float* __restrict__ x1, const float* __restrict__ x2,
    const float* __restrict__ fe1, const float* __restrict__ fe2,
    const float* __restrict__ G1, const float* __restrict__ G2,
    unsigned short* __restrict__ REB) {
  __shared__ float g1s[512];
  __shared__ float g2s[512];
  const int n  = blockIdx.y;
  const int hw = blockIdx.x * 256 + threadIdx.x;
  for (int i = threadIdx.x; i < 512; i += 256) {
    g1s[i] = G1[(size_t)n * 512 + i];
    g2s[i] = G2[(size_t)n * 512 + i];
  }
  __syncthreads();

  const size_t base = (size_t)n * 512 * HWSZ + hw;
  const float* p1 = x1 + base;
  const float* p2 = x2 + base;

  float m1 = -1e30f, z1 = 0.f, w1 = 0.f;
  float m2 = -1e30f, z2 = 0.f, w2 = 0.f;
  for (int c = 0; c < 512; ++c) {
    const float a = p1[(size_t)c * HWSZ];
    const float b = p2[(size_t)c * HWSZ];
    const float r1 = fmaf(g1s[c], a, b);     // weight_x1 + x2
    const float r2 = fmaf(g2s[c], b, a);     // weight_x2 + x1
    float t = expf(r1);
    if (t > m1) { const float s = expf(m1 - t); z1 = z1 * s + 1.f; w1 = w1 * s + r1; m1 = t; }
    else        { const float e = expf(t - m1); z1 += e; w1 += e * r1; }
    t = expf(r2);
    if (t > m2) { const float s = expf(m2 - t); z2 = z2 * s + 1.f; w2 = w2 * s + r2; m2 = t; }
    else        { const float e = expf(t - m2); z2 += e; w2 += e * r2; }
  }
  const float s1 = w1 / z1;                  // spatial gate scalar, branch 1
  const float s2 = w2 / z2;

  const float* q1 = fe1 + base;
  const float* q2 = fe2 + base;
  unsigned short* o1 = REB + (size_t)n * KDIM * HWSZ + hw;           // c in [0,512)
  unsigned short* o2 = o1 + (size_t)512 * HWSZ;                      // c in [512,1024)
  for (int c = 0; c < 512; ++c) {
    const float a = p1[(size_t)c * HWSZ];
    const float b = p2[(size_t)c * HWSZ];
    const float co1 = fmaf(a, s1, b) + q1[(size_t)c * HWSZ];         // x1*s1 + x2 + FE1
    const float co2 = fmaf(b, s2, a) + q2[(size_t)c * HWSZ];         // x2*s2 + x1 + FE2
    o1[(size_t)c * HWSZ] = f2bf(co1);
    o2[(size_t)c * HWSZ] = f2bf(co2);
  }
}

// ---------------------------------------------------------------------------
extern "C" void kernel_launch(void* const* d_in, const int* in_sizes, int n_in,
                              void* d_out, int out_size, void* d_ws, size_t ws_size,
                              hipStream_t stream) {
  const float* x1    = (const float*)d_in[0];
  const float* x2    = (const float*)d_in[1];
  const float* fe1   = (const float*)d_in[2];
  const float* fe2   = (const float*)d_in[3];
  const float* c1_w  = (const float*)d_in[4];
  const float* c1_b  = (const float*)d_in[5];
  const float* c2_w  = (const float*)d_in[6];
  const float* c2_b  = (const float*)d_in[7];
  const float* m1_w1 = (const float*)d_in[8];
  const float* m1_b1 = (const float*)d_in[9];
  const float* m1_w2 = (const float*)d_in[10];
  const float* m1_b2 = (const float*)d_in[11];
  const float* m2_w1 = (const float*)d_in[12];
  const float* m2_b1 = (const float*)d_in[13];
  const float* m2_w2 = (const float*)d_in[14];
  const float* m2_b2 = (const float*)d_in[15];
  const float* p1_w  = (const float*)d_in[16];
  const float* p1_b  = (const float*)d_in[17];
  const float* p2_w  = (const float*)d_in[18];
  const float* p2_b  = (const float*)d_in[19];

  float* wsf = (float*)d_ws;
  float* X1X = wsf;                          // 16777216
  float* X2X = wsf + 16777216;               // 16777216
  float* P1  = wsf + 33554432;               // 16384
  float* P2  = wsf + 33570816;
  float* G1  = wsf + 33587200;
  float* G2  = wsf + 33603584;
  unsigned short* REB = (unsigned short*)(wsf + 33619968);  // 32*1024*1024 bf16

  float* po1 = (float*)d_out;
  float* po2 = po1 + 16777216;

  // 1) x1x / x2x = conv1x1(concat(x1,x2), c1/c2)   [WMMA bf16]
  gemm_dual_kernel<0><<<dim3(8, 8, NB), 256, 0, stream>>>(
      c1_w, c1_b, X1X, c2_w, c2_b, X2X, x1, x2, nullptr);

  // 2) softmax-of-exp pooling over HW
  pool_kernel<<<dim3(2048, 2), 256, 0, stream>>>(X1X, X2X, P1, P2);

  // 3) sigmoid MLP channel gates
  mlp_gate_kernel<<<dim3(NB, 2), 256, 0, stream>>>(
      P1, P2, m1_w1, m1_b1, m1_w2, m1_b2, m2_w1, m2_b1, m2_w2, m2_b2, G1, G2);

  // 4) channel gating + spatial softmax gates + residuals -> RE (bf16)
  fuse_spatial_kernel<<<dim3(4, NB), 256, 0, stream>>>(
      x1, x2, fe1, fe2, G1, G2, REB);

  // 5) po_x1 / po_x2 = conv1x1(RE, p1/p2)          [WMMA bf16]
  gemm_dual_kernel<1><<<dim3(8, 8, NB), 256, 0, stream>>>(
      p1_w, p1_b, po1, p2_w, p2_b, po2, nullptr, nullptr, REB);
}
